// NonLocalBlock_19035295056489
// MI455X (gfx1250) — matrix operand order repaired and checked by reading, not compiled, not run on hardware
//
#include <hip/hip_runtime.h>

// ---------------------------------------------------------------------------
// CDNA5 (gfx1250) NonLocalBlock: GroupNorm + QKV GEMMs + flash attention +
// output projection. Matrix math: v_wmma_f32_16x16x32_bf16 (wave32).
// Attention K/V/Q tile staging: Tensor Data Mover (tensor_load_to_lds) with
// TENSORcnt-based software pipelining (single-buffer, phase-offset K vs V).
// ---------------------------------------------------------------------------

typedef __attribute__((ext_vector_type(16))) __bf16 v16bf;
typedef __attribute__((ext_vector_type(8)))  __bf16 v8bf;
typedef __attribute__((ext_vector_type(4)))  __bf16 v4bf;
typedef __attribute__((ext_vector_type(8)))  float  v8f;
typedef __attribute__((ext_vector_type(4)))  unsigned int v4u;
typedef __attribute__((ext_vector_type(8)))  int v8i;
typedef __attribute__((ext_vector_type(4)))  int v4i;

#define DEV __device__ __forceinline__

// A fragment (16x32 bf16, row-major LDS tile with leading dim lda):
// lane L: M = L%16, h = L/16; elements 0..7 = K(8h..8h+7), 8..15 = K(16+8h..).
DEV v16bf frag_a(const __bf16* t, int lda, int lane) {
  const int m = lane & 15, h = (lane >> 4) & 1;
  const __bf16* p = t + m * lda + 8 * h;
  v8bf lo = *(const v8bf*)(p);
  v8bf hi = *(const v8bf*)(p + 16);
  return __builtin_shufflevector(lo, hi, 0,1,2,3,4,5,6,7,8,9,10,11,12,13,14,15);
}

// B fragment (32x16 bf16). LDS tile is "n-major": element (k,n) at n*ldb + k.
// lane L: N = L%16, h = L/16; elements 0..7 = K(8h..8h+7), 8..15 = K(16+8h..).
DEV v16bf frag_b(const __bf16* t, int ldb, int lane) {
  const int n = lane & 15, h = (lane >> 4) & 1;
  const __bf16* p = t + n * ldb + 8 * h;
  v8bf lo = *(const v8bf*)(p);
  v8bf hi = *(const v8bf*)(p + 16);
  return __builtin_shufflevector(lo, hi, 0,1,2,3,4,5,6,7,8,9,10,11,12,13,14,15);
}

DEV v8f wmma_bf16(v16bf a, v16bf b, v8f c) {
  return __builtin_amdgcn_wmma_f32_16x16x32_bf16(
      /*neg_a=*/false, a, /*neg_b=*/false, b,
      /*c_mod=*/(short)0, c, /*reuse_a=*/false, /*reuse_b=*/false);
}

// ---------------------------------------------------------------------------
// Tensor Data Mover: 2D tile load global -> LDS, with LDS row padding.
// D# per CDNA5 ISA 8.3/8.4. data_size=3 (8-byte units). count=1, type=2.
// pad_interval code c -> pad after 2^(c+1) DWORDs; pad_amount code p -> p+1 DW.
// (amdgpu-toolchain 6-arg builtin form.)
// ---------------------------------------------------------------------------
DEV void tdm_load_2d(unsigned lds_addr, const void* gaddr,
                     unsigned tile_w_u,    // tile width, 8B units
                     unsigned tile_rows,   // tile height (rows)
                     unsigned tensor_w_u,  // tensor row length, 8B units
                     unsigned tensor_rows, // tensor height (rows)
                     unsigned row_stride_u,// row stride, 8B units
                     unsigned pad_int_code, unsigned pad_amt_code) {
  unsigned long ga = (unsigned long)gaddr;
  v4u g0;
  g0[0] = 1u;                                        // count=1, user mode
  g0[1] = lds_addr;                                  // LDS byte address
  g0[2] = (unsigned)ga;                              // global addr [95:64]
  g0[3] = (unsigned)((ga >> 32) & 0x1ffffffu) | (2u << 30);  // [120:96]|type=2
  unsigned long w0 = (unsigned long)((3u << 16) |    // data_size = 8B
                                     (1u << 20) |    // pad_enable
                                     (pad_int_code << 22) |
                                     (pad_amt_code << 25)) |
                     ((unsigned long)(tensor_w_u & 0xffffu) << 48);
  unsigned long w1 = (unsigned long)((tensor_w_u >> 16) & 0xffffu) |
                     ((unsigned long)tensor_rows << 16) |
                     ((unsigned long)(tile_w_u & 0xffffu) << 48);
  unsigned long w2 = (unsigned long)(tile_rows & 0xffffu) |
                     ((unsigned long)row_stride_u << 32);   // stride[31:0]
  unsigned long w3 = 0;                                     // stride[47:32]=0
  v8i g1;
  g1[0] = (int)w0; g1[1] = (int)(w0 >> 32);
  g1[2] = (int)w1; g1[3] = (int)(w1 >> 32);
  g1[4] = (int)w2; g1[5] = (int)(w2 >> 32);
  g1[6] = (int)w3; g1[7] = (int)(w3 >> 32);
  v4i z4 = {0, 0, 0, 0};
  v8i z8 = {0, 0, 0, 0, 0, 0, 0, 0};
  __builtin_amdgcn_tensor_load_to_lds(g0, g1, z4, z4, z8, 0);
}

DEV unsigned lds_off32(const void* p) {  // flat LDS address: offset in [31:0]
  return (unsigned)(unsigned long)p;
}

// ---------------------------------------------------------------------------
// Weight fp32 -> bf16 convert
// ---------------------------------------------------------------------------
__global__ __launch_bounds__(256) void cvt_f32_bf16(
    const float* __restrict__ s, __bf16* __restrict__ d, int n) {
  int i = blockIdx.x * 256 + threadIdx.x;
  if (i < n) d[i] = (__bf16)s[i];
}

// ---------------------------------------------------------------------------
// GroupNorm over (b, 32 groups of 16ch, 4096 pix); writes h^T (pix, C) bf16.
// ---------------------------------------------------------------------------
__global__ __launch_bounds__(256) void groupnorm_to_bf16T(
    const float* __restrict__ x, const float* __restrict__ gamma,
    const float* __restrict__ beta, __bf16* __restrict__ hT,
    int C, int HW) {
  const int G = 32;
  const int b = blockIdx.x / G, g = blockIdx.x % G;
  const long base = (long)b * C * HW + (long)g * 16 * HW;
  const int tid = threadIdx.x;
  const int NE = 16 * HW;
  float s = 0.f, ss = 0.f;
  for (int i = tid; i < NE; i += 256) {
    float v = x[base + i];
    s += v; ss += v * v;
  }
  __shared__ float r1[256], r2[256];
  r1[tid] = s; r2[tid] = ss;
  __syncthreads();
  for (int off = 128; off > 0; off >>= 1) {
    if (tid < off) { r1[tid] += r1[tid + off]; r2[tid] += r2[tid + off]; }
    __syncthreads();
  }
  const float mean = r1[0] / (float)NE;
  const float var  = r2[0] / (float)NE - mean * mean;
  const float rstd = rsqrtf(var + 1e-6f);
  const int ch = tid & 15;
  const float gm = gamma[g * 16 + ch], bt = beta[g * 16 + ch];
  for (int pix = tid >> 4; pix < HW; pix += 16) {
    float v = x[base + (long)ch * HW + pix];
    hT[((long)b * HW + pix) * C + g * 16 + ch] = (__bf16)((v - mean) * rstd * gm + bt);
  }
}

// ---------------------------------------------------------------------------
// WMMA GEMM:  D(M,N) = A(M,K) @ B(N,K)^T  (+ bias + residual), bf16 in, f32 acc
// BIAS_MODE: 0=none, 1=bias[col], 2=bias[row]. Block tile 128x64, 8 waves.
// ---------------------------------------------------------------------------
template <int BIAS_MODE, bool OUT_BF16, bool ADD_RES>
__global__ __launch_bounds__(256) void wmma_gemm_nt(
    const __bf16* __restrict__ A, long sA, int lda,
    const __bf16* __restrict__ Bm, long sB, int ldb,
    const float* __restrict__ bias,
    const float* __restrict__ res, long sRes, int ldr,
    void* __restrict__ outp, long sOut, int ldo,
    int M, int N, int K) {
  constexpr int BM = 128, BN = 64, KT = 32, LDT = 40;
  __shared__ alignas(16) __bf16 As[BM * LDT];
  __shared__ alignas(16) __bf16 Bs[BN * LDT];
  const int b = blockIdx.z;
  const int m0 = blockIdx.x * BM;
  const int n0 = blockIdx.y * BN;
  A  += (long)b * sA;
  Bm += (long)b * sB;
  const int tid = threadIdx.x, lane = tid & 31, w = tid >> 5;
  const int wm = w & 3;
  const int wn = w >> 2;
  v8f acc[2][2] = {};

  for (int k0 = 0; k0 < K; k0 += KT) {
    __syncthreads();
#pragma unroll
    for (int ii = 0; ii < BM * (KT / 4) / 256; ++ii) {
      int i = tid + ii * 256;
      int r = i >> 3, c = (i & 7) << 2;
      *(v4bf*)(As + r * LDT + c) = *(const v4bf*)(A + (long)(m0 + r) * lda + k0 + c);
    }
#pragma unroll
    for (int ii = 0; ii < BN * (KT / 4) / 256; ++ii) {
      int i = tid + ii * 256;
      int r = i >> 3, c = (i & 7) << 2;
      *(v4bf*)(Bs + r * LDT + c) = *(const v4bf*)(Bm + (long)(n0 + r) * ldb + k0 + c);
    }
    if (k0 + KT < K) {  // prefetch next K tile (64B line = full tile row)
      if (tid < BM)            __builtin_prefetch(A  + (long)(m0 + tid) * lda + k0 + KT, 0, 3);
      else if (tid < BM + BN)  __builtin_prefetch(Bm + (long)(n0 + tid - BM) * ldb + k0 + KT, 0, 3);
    }
    __syncthreads();
    v16bf a0 = frag_a(As + (wm * 32 + 0)  * LDT, LDT, lane);
    v16bf a1 = frag_a(As + (wm * 32 + 16) * LDT, LDT, lane);
    v16bf b0 = frag_b(Bs + (wn * 32 + 0)  * LDT, LDT, lane);
    v16bf b1 = frag_b(Bs + (wn * 32 + 16) * LDT, LDT, lane);
    acc[0][0] = wmma_bf16(a0, b0, acc[0][0]);
    acc[0][1] = wmma_bf16(a0, b1, acc[0][1]);
    acc[1][0] = wmma_bf16(a1, b0, acc[1][0]);
    acc[1][1] = wmma_bf16(a1, b1, acc[1][1]);
  }

  const int h = (lane >> 4) & 1, nl = lane & 15;
#pragma unroll
  for (int im = 0; im < 2; ++im)
#pragma unroll
    for (int in = 0; in < 2; ++in)
#pragma unroll
      for (int r = 0; r < 8; ++r) {
        int row = m0 + wm * 32 + im * 16 + r + 8 * h;
        int col = n0 + wn * 32 + in * 16 + nl;
        float v = acc[im][in][r];
        if (BIAS_MODE == 1) v += bias[col];
        if (BIAS_MODE == 2) v += bias[row];
        if (ADD_RES) v += res[(long)b * sRes + (long)row * ldr + col];
        if (OUT_BF16)
          ((__bf16*)outp)[(long)b * sOut + (long)row * ldo + col] = (__bf16)v;
        else
          ((float*)outp)[(long)b * sOut + (long)row * ldo + col] = v;
      }
}

// ---------------------------------------------------------------------------
// Flash attention. WG = 32 query rows, 8 waves; key blocks of 64.
// Wave w: rt=w>>2 (query row tile), cq=w&3 (S key-subtile / PV 128-ch slice).
// Tile staging via TDM, pipelined with TENSORcnt (wave0 issues+waits):
//   K_{i+1} issued after S_i (overlaps softmax+PV_i);
//   V_{i+1} issued after PV_i (overlaps S_{i+1}).
// LDS padding done by TDM pad_interval/pad_amount (row strides 520 / 72).
// ---------------------------------------------------------------------------
__global__ __launch_bounds__(256) void flash_attn(
    const __bf16* __restrict__ qT, const __bf16* __restrict__ kT,
    const __bf16* __restrict__ vCN, __bf16* __restrict__ aT,
    int HW, int C, float scale) {
  constexpr int LDQ = 520, LDK = 520, LDV = 72, LDSS = 65, LDP = 72;
  constexpr int QB = 32, KB = 64;
  extern __shared__ char smem[];
  float*  Sbuf = (float*)(smem);            //  8320 B
  float*  Abuf = (float*)(smem + 8320);     //   128 B
  float*  Lbuf = (float*)(smem + 8448);     //   128 B
  __bf16* Qs   = (__bf16*)(smem + 8576);    // 33280 B : 32 x 520 (TDM pad)
  __bf16* Ks   = (__bf16*)(smem + 41856);   // 66560 B : 64 x 520 (TDM pad)
  __bf16* Ps   = (__bf16*)(smem + 108416);  //  4608 B : 32 x 72
  __bf16* Vs   = (__bf16*)(smem + 113024);  // 73728 B : 512 x 72 (TDM pad)

  const int b = blockIdx.y;
  const int q0 = blockIdx.x * QB;
  const int tid = threadIdx.x, lane = tid & 31, w = tid >> 5;
  const int rt = w >> 2;
  const int cq = w & 3;
  const __bf16* qg = qT  + ((long)b * HW + q0) * C;
  const __bf16* kg = kT  + (long)b * HW * C;
  const __bf16* vg = vCN + (long)b * C * HW;   // channel-major (C, HW)

  const unsigned CW = (unsigned)(C * 2 / 8);   // 512ch row = 128 x 8B units
  const unsigned VW = (unsigned)(KB * 2 / 8);  // 64-key row = 16 x 8B units
  // Prologue: Q (32x512), K_0 (64x512), V_0 (512x64). Wait so that Q,K0 are
  // done and V0 may remain in flight (matches steady-state invariant).
  if (w == 0) {
    tdm_load_2d(lds_off32(Qs), qg,      CW, QB, CW, (unsigned)QB, CW, 7, 3);
    tdm_load_2d(lds_off32(Ks), kg,      CW, KB, CW, (unsigned)HW, CW, 7, 3);
    tdm_load_2d(lds_off32(Vs), vg,      VW, (unsigned)C, (unsigned)(HW * 2 / 8),
                (unsigned)C, (unsigned)(HW * 2 / 8), 4, 3);
    __builtin_amdgcn_s_wait_tensorcnt((short)1);
  }
  __syncthreads();

  float m_i = -3.0e38f, l_i = 0.f;  // wave0: lane = row
  v8f oacc[8] = {};                 // 16 rows x 128 channels

  for (int kb = 0; kb < HW; kb += KB) {
    const bool more = (kb + KB < HW);
    // ---- S subtile: 16 queries (rt) x 16 keys (cq) over C=512 ----
    v8f s = {};
#pragma unroll
    for (int ks = 0; ks < 16; ++ks) {
      v16bf a  = frag_a(Qs + rt * 16 * LDQ + ks * 32, LDQ, lane);
      v16bf bb = frag_b(Ks + cq * 16 * LDK + ks * 32, LDK, lane);
      s = wmma_bf16(a, bb, s);
    }
    {
      const int h = (lane >> 4) & 1, nl = lane & 15;
#pragma unroll
      for (int r = 0; r < 8; ++r)
        Sbuf[(rt * 16 + r + 8 * h) * LDSS + cq * 16 + nl] = s[r];
    }
    __syncthreads();  // Ks free from here

    // ---- wave0: prefetch K_{i+1}; ensure V_i arrived; online softmax ----
    if (w == 0) {
      if (more) {
        tdm_load_2d(lds_off32(Ks), kg + (long)(kb + KB) * C,
                    CW, KB, CW, (unsigned)HW, CW, 7, 3);
        __builtin_amdgcn_s_wait_tensorcnt((short)1);  // V_i done, K_next flies
      } else {
        __builtin_amdgcn_s_wait_tensorcnt((short)0);  // V_i done
      }
      const int row = lane;
      float bm = -3.0e38f;
#pragma unroll 8
      for (int j = 0; j < KB; ++j) bm = fmaxf(bm, Sbuf[row * LDSS + j]);
      bm *= scale;
      float mnew = fmaxf(m_i, bm);
      float al = __expf(m_i - mnew);
      float sum = 0.f;
#pragma unroll 8
      for (int j = 0; j < KB; ++j) {
        float p = __expf(Sbuf[row * LDSS + j] * scale - mnew);
        Ps[row * LDP + j] = (__bf16)p;
        sum += p;
      }
      l_i = l_i * al + sum;
      m_i = mnew;
      Abuf[row] = al;
    }
    __syncthreads();

    // ---- rescale accumulators, then O += P(16x64) @ V^T(64x128 slice) ----
    {
      const int h = (lane >> 4) & 1;
#pragma unroll
      for (int r = 0; r < 8; ++r) {
        float al = Abuf[rt * 16 + r + 8 * h];
#pragma unroll
        for (int ct = 0; ct < 8; ++ct) oacc[ct][r] *= al;
      }
    }
#pragma unroll
    for (int ks = 0; ks < KB / 32; ++ks) {
      v16bf a = frag_a(Ps + rt * 16 * LDP + ks * 32, LDP, lane);
#pragma unroll
      for (int ct = 0; ct < 8; ++ct) {
        v16bf bb = frag_b(Vs + (cq * 128 + ct * 16) * LDV + ks * 32, LDV, lane);
        oacc[ct] = wmma_bf16(a, bb, oacc[ct]);
      }
    }
    __syncthreads();  // Vs free from here

    // ---- wave0: prefetch V_{i+1}; ensure K_{i+1} arrived ----
    if (w == 0) {
      if (more) {
        tdm_load_2d(lds_off32(Vs), vg + (kb + KB),
                    VW, (unsigned)C, (unsigned)(HW * 2 / 8),
                    (unsigned)C, (unsigned)(HW * 2 / 8), 4, 3);
        __builtin_amdgcn_s_wait_tensorcnt((short)1);  // K_next done, V flies
      }
    }
    __syncthreads();
  }

  __syncthreads();
  if (w == 0) Lbuf[lane] = l_i;
  __syncthreads();
  {
    const int h = (lane >> 4) & 1, nl = lane & 15;
#pragma unroll
    for (int r = 0; r < 8; ++r) {
      const int row = rt * 16 + r + 8 * h;
      const float inv = 1.f / Lbuf[row];
      const long qrow = (long)b * HW + q0 + row;
#pragma unroll
      for (int ct = 0; ct < 8; ++ct) {
        int ch = cq * 128 + ct * 16 + nl;
        aT[qrow * C + ch] = (__bf16)(oacc[ct][r] * inv);
      }
    }
  }
}

// ---------------------------------------------------------------------------
// Host-side orchestration
// ---------------------------------------------------------------------------
extern "C" void kernel_launch(void* const* d_in, const int* in_sizes, int n_in,
                              void* d_out, int out_size, void* d_ws, size_t ws_size,
                              hipStream_t stream) {
  const float* x    = (const float*)d_in[0];
  const float* gn_w = (const float*)d_in[1];
  const float* gn_b = (const float*)d_in[2];
  const float* wq   = (const float*)d_in[3];
  const float* bq   = (const float*)d_in[4];
  const float* wk   = (const float*)d_in[5];
  const float* bk   = (const float*)d_in[6];
  const float* wv   = (const float*)d_in[7];
  const float* bv   = (const float*)d_in[8];
  const float* wp   = (const float*)d_in[9];
  const float* bp   = (const float*)d_in[10];

  const int B = 4, C = 512, HW = 4096, G = 32;
  const long PT = (long)B * HW * C;

  __bf16* hT  = (__bf16*)d_ws;      // (b, pix, C)
  __bf16* qTp = hT  + PT;           // (b, pix, C)
  __bf16* kTp = qTp + PT;           // (b, pix, C)
  __bf16* vCN = kTp + PT;           // (b, C, pix)  channel-major for TDM tiles
  __bf16* aTp = vCN + PT;           // (b, pix, C)
  __bf16* wqb = aTp + PT;
  __bf16* wkb = wqb + (long)C * C;
  __bf16* wvb = wkb + (long)C * C;
  __bf16* wpb = wvb + (long)C * C;

  const int WN = C * C;
  cvt_f32_bf16<<<WN / 256, 256, 0, stream>>>(wq, wqb, WN);
  cvt_f32_bf16<<<WN / 256, 256, 0, stream>>>(wk, wkb, WN);
  cvt_f32_bf16<<<WN / 256, 256, 0, stream>>>(wv, wvb, WN);
  cvt_f32_bf16<<<WN / 256, 256, 0, stream>>>(wp, wpb, WN);

  groupnorm_to_bf16T<<<B * G, 256, 0, stream>>>(x, gn_w, gn_b, hT, C, HW);

  dim3 blk(256);
  // q/k: D(pix,o) = hT(pix,c) @ W(o,c)^T + b[o]  (pixel-major bf16)
  dim3 gq(HW / 128, C / 64, B);
  wmma_gemm_nt<1, true, false><<<gq, blk, 0, stream>>>(
      hT, (long)HW * C, C, wqb, 0L, C, bq, nullptr, 0L, 0,
      qTp, (long)HW * C, C, HW, C, C);
  wmma_gemm_nt<1, true, false><<<gq, blk, 0, stream>>>(
      hT, (long)HW * C, C, wkb, 0L, C, bk, nullptr, 0L, 0,
      kTp, (long)HW * C, C, HW, C, C);
  // v: D(o,pix) = wv(o,c) @ hT(pix,c)^T + bv[o]  (channel-major bf16)
  dim3 gv(C / 128, HW / 64, B);
  wmma_gemm_nt<2, true, false><<<gv, blk, 0, stream>>>(
      wvb, 0L, C, hT, (long)HW * C, C, bv, nullptr, 0L, 0,
      vCN, (long)C * HW, HW, C, HW, C);

  // flash attention (dynamic LDS: 186752 B, fits 320KB/WGP)
  const size_t shmem = 186752;
  flash_attn<<<dim3(HW / 32, B), blk, shmem, stream>>>(
      qTp, kTp, vCN, aTp, HW, C, 0.044194173824159216f /* 512^-0.5 */);

  // projection: out(o,pix) = wp(o,c) @ aT(pix,c)^T + bp[o] + x   (f32 out)
  dim3 gp(C / 128, HW / 64, B);
  wmma_gemm_nt<2, false, true><<<gp, blk, 0, stream>>>(
      wpb, 0L, C, aTp, (long)HW * C, C, bp, x, (long)C * HW, HW,
      d_out, (long)C * HW, HW, C, HW, C);
}
